// GraphAttentionLayer_46127948759743
// MI455X (gfx1250) — matrix operand order repaired and checked by reading, not compiled, not run on hardware
//
#include <hip/hip_runtime.h>

// ---------------------------------------------------------------------------
// GAT layer for MI455X (gfx1250, wave32, WMMA).
//   C=32 cells, N=8192 genes, IN_F=256, OUT_F=128.
//   Pipeline: [memset hp] -> prep Wt(f16,transposed) -> WMMA GEMM (hW)
//             -> per-row scores s1/s2 -> masked softmax (att) -> h_prime.
// GEMM v3: whole A panel (32 rows x 256 K, f16 = 128 VGPRs) preloaded and
//          converted once per wave; nt-outer/ks-inner loop with software-
//          pipelined ds_load_b128 B fragments from LDS-staged Wt. Removes
//          the per-tile s_wait_dscnt-0 stall and WMMA->VALU cvt hazard nops.
// Workspace layout (bytes):
//   [0, 64K)              : Wt  = W^T in f16, 128 x 256 (row n, col k)
//   [64K, 64K+128M)       : hW  fp32, (C*N) x 128   (mostly L2-resident)
//   then                  : s1  fp32, C*N
//   then                  : s2  fp32, C*N
// ---------------------------------------------------------------------------

typedef __attribute__((ext_vector_type(16))) _Float16 v16h;
typedef __attribute__((ext_vector_type(8)))  _Float16 v8h;
typedef __attribute__((ext_vector_type(8)))  float    v8f;
typedef __attribute__((ext_vector_type(4)))  float    v4f;

#define NCELLS 32
#define NGENES 8192
#define IN_F   256
#define OUT_F  128
#define ALPHA  0.2f
#define NEG_INF -9.0e15f

// ---------------------------------------------------------------- Wt prep ---
// Wt[n][k] = (f16) W[k][n];  32768 elements total.
__global__ __launch_bounds__(256)
void gat_prep_wt(const float* __restrict__ W, _Float16* __restrict__ Wt) {
    int t = blockIdx.x * 256 + threadIdx.x;   // [0, 32768)
    int n = t >> 8;                           // [0, 128)
    int k = t & 255;                          // [0, 256)
    Wt[n * IN_F + k] = (_Float16)W[k * OUT_F + n];
}

// ------------------------------------------------------------------- GEMM ---
// Grid: NCELLS*32 blocks, 256 threads (8 waves). Block: 256 rows x 128 cols.
// Each wave: 32 rows (two 16-row M-tiles sharing every B fragment).
__global__ __launch_bounds__(256)
void gat_gemm(const float* __restrict__ x, const _Float16* __restrict__ Wt,
              float* __restrict__ hW) {
    extern __shared__ _Float16 Wl[];        // 128 x 256 f16 = 64 KB

    const int cell   = blockIdx.x >> 5;     // 32 row-blocks of 256 per cell
    const int rowblk = blockIdx.x & 31;
    const int wave   = threadIdx.x >> 5;
    const int lane   = threadIdx.x & 31;
    const int lhalf  = lane >> 4;           // 0: lanes 0-15, 1: lanes 16-31
    const int l15    = lane & 15;

    // ---- Cooperative Wt -> LDS copy: 64 KB, 256 B per thread (16 x b128).
    {
        const v8h* src = (const v8h*)Wt;    // 16-byte chunks, 4096 total
        v8h* dst = (v8h*)Wl;
#pragma unroll
        for (int i = 0; i < 16; i++) {
            const int chunk = i * 256 + threadIdx.x;
            dst[chunk] = src[chunk];
        }
    }

    const int m0 = rowblk * 256 + wave * 32;            // wave's 32-row strip
    // A rows for this lane (ISA A-layout: M = lane&15, same row both halves)
    const float* aptr0 = x + (long)(cell * NGENES + m0 + l15) * IN_F;
    const float* aptr1 = aptr0 + 16 * IN_F;

    // Speculative prefetch of the next block's A data (dropped if OOB).
    __builtin_prefetch(aptr0 + 256 * IN_F, 0, 1);

    // ---- Preload + convert the wave's entire A panel: 2 M-tiles x 8 K-steps.
    // Fragment layout per ISA: lanes 0-15 carry K=kk+[0..7]u[16..23],
    //                          lanes 16-31 carry K=kk+[8..15]u[24..31].
    v16h A0[8], A1[8];                                  // 128 VGPRs
#pragma unroll
    for (int ks = 0; ks < 8; ks++) {
        const int kb = ks * 32 + lhalf * 8;
        v4f p0 = *(const v4f*)(aptr0 + kb);
        v4f p1 = *(const v4f*)(aptr0 + kb + 4);
        v4f p2 = *(const v4f*)(aptr0 + kb + 16);
        v4f p3 = *(const v4f*)(aptr0 + kb + 20);
        v4f q0 = *(const v4f*)(aptr1 + kb);
        v4f q1 = *(const v4f*)(aptr1 + kb + 4);
        v4f q2 = *(const v4f*)(aptr1 + kb + 16);
        v4f q3 = *(const v4f*)(aptr1 + kb + 20);
#pragma unroll
        for (int i = 0; i < 4; i++) {
            A0[ks][i]      = (_Float16)p0[i];
            A0[ks][i + 4]  = (_Float16)p1[i];
            A0[ks][i + 8]  = (_Float16)p2[i];
            A0[ks][i + 12] = (_Float16)p3[i];
            A1[ks][i]      = (_Float16)q0[i];
            A1[ks][i + 4]  = (_Float16)q1[i];
            A1[ks][i + 8]  = (_Float16)q2[i];
            A1[ks][i + 12] = (_Float16)q3[i];
        }
    }

    __syncthreads();                        // Wt fully staged in LDS

    // B fragment base for this lane: N-row (l15) of Wt, K-halves by lane half.
    const _Float16* bbase = Wl + l15 * IN_F + lhalf * 8;
    float* outbase = hW + (long)(cell * NGENES) * OUT_F;

#pragma unroll
    for (int nt = 0; nt < 8; nt++) {
        const _Float16* bnt = bbase + (nt * 16) * IN_F;

        // Preload B fragment for ks=0.
        v16h bcur;
        {
            v8h t0 = *(const v8h*)(bnt);
            v8h t1 = *(const v8h*)(bnt + 16);
#pragma unroll
            for (int i = 0; i < 8; i++) { bcur[i] = t0[i]; bcur[i + 8] = t1[i]; }
        }

        v8f c0 = {}, c1 = {};
#pragma unroll
        for (int ks = 0; ks < 8; ks++) {
            // Software pipeline: issue next B-fragment load before this
            // step's WMMAs so the ds wait never lands at depth 0.
            v16h bnext = bcur;
            if (ks < 7) {
                const _Float16* p = bnt + (ks + 1) * 32;
                v8h t0 = *(const v8h*)(p);
                v8h t1 = *(const v8h*)(p + 16);
#pragma unroll
                for (int i = 0; i < 8; i++) { bnext[i] = t0[i]; bnext[i + 8] = t1[i]; }
            }
            c0 = __builtin_amdgcn_wmma_f32_16x16x32_f16(
                false, A0[ks], false, bcur, (short)0, c0, false, false);
            c1 = __builtin_amdgcn_wmma_f32_16x16x32_f16(
                false, A1[ks], false, bcur, (short)0, c1, false, false);
            bcur = bnext;
        }

        // ---- Store this N-tile (overlaps with next tile's WMMAs).
        // C/D layout: VGPR r, lane l -> M = m0 + r + 8*(l>>4), N = nt*16+(l&15)
        const int n = nt * 16 + l15;
#pragma unroll
        for (int r = 0; r < 8; r++) {
            const int m = m0 + r + lhalf * 8;
            outbase[(long)m * OUT_F + n]        = c0[r];
            outbase[(long)(m + 16) * OUT_F + n] = c1[r];
        }
    }
}

// ----------------------------------------------------------------- scores ---
// One wave per row: s1[row] = hW[row,:]·a1[c,:], s2[row] = hW[row,:]·a2[c,:]
__global__ __launch_bounds__(256)
void gat_scores(const float* __restrict__ hW, const float* __restrict__ a,
                float* __restrict__ s1, float* __restrict__ s2) {
    const int wave = threadIdx.x >> 5;
    const int lane = threadIdx.x & 31;
    const int row  = blockIdx.x * 8 + wave;        // [0, NCELLS*NGENES)
    const int cell = row >> 13;                    // row / NGENES

    v4f h  = *(const v4f*)(hW + (long)row * OUT_F + lane * 4);
    v4f a1 = *(const v4f*)(a + cell * 2 * OUT_F + lane * 4);
    v4f a2 = *(const v4f*)(a + cell * 2 * OUT_F + OUT_F + lane * 4);

    float p1 = h[0]*a1[0] + h[1]*a1[1] + h[2]*a1[2] + h[3]*a1[3];
    float p2 = h[0]*a2[0] + h[1]*a2[1] + h[2]*a2[2] + h[3]*a2[3];
#pragma unroll
    for (int off = 16; off > 0; off >>= 1) {
        p1 += __shfl_xor(p1, off, 32);
        p2 += __shfl_xor(p2, off, 32);
    }
    if (lane == 0) { s1[row] = p1; s2[row] = p2; }
}

// ---------------------------------------------------------------- softmax ---
// One block per cell. e[j] = lrelu(s1[(2j)%N] + s2[(2j+1)%N]); mask by adj;
// numerically stable softmax over N (matches JAX incl. all-masked case).
__global__ __launch_bounds__(1024)
void gat_softmax(const float* __restrict__ s1, const float* __restrict__ s2,
                 const int* __restrict__ adj, float* __restrict__ att) {
    const int c    = blockIdx.x;
    const int t    = threadIdx.x;
    const int lane = t & 31;
    const int wv   = t >> 5;
    __shared__ float red[32];

    float v[8];
    float lmax = -3.0e38f;
#pragma unroll
    for (int i = 0; i < 8; i++) {
        const int j  = i * 1024 + t;
        const int i1 = (2 * j) & (NGENES - 1);
        const int i2 = (2 * j + 1) & (NGENES - 1);
        float e = s1[c * NGENES + i1] + s2[c * NGENES + i2];
        e = e > 0.0f ? e : ALPHA * e;
        v[i] = (adj[c * NGENES + j] > 0) ? e : NEG_INF;
        lmax = fmaxf(lmax, v[i]);
    }
#pragma unroll
    for (int off = 16; off > 0; off >>= 1) lmax = fmaxf(lmax, __shfl_xor(lmax, off, 32));
    if (lane == 0) red[wv] = lmax;
    __syncthreads();
    if (wv == 0) {
        float m = red[lane];
#pragma unroll
        for (int off = 16; off > 0; off >>= 1) m = fmaxf(m, __shfl_xor(m, off, 32));
        if (lane == 0) red[0] = m;
    }
    __syncthreads();
    const float m = red[0];
    __syncthreads();

    float lsum = 0.0f;
#pragma unroll
    for (int i = 0; i < 8; i++) { v[i] = expf(v[i] - m); lsum += v[i]; }
#pragma unroll
    for (int off = 16; off > 0; off >>= 1) lsum += __shfl_xor(lsum, off, 32);
    if (lane == 0) red[wv] = lsum;
    __syncthreads();
    if (wv == 0) {
        float s = red[lane];
#pragma unroll
        for (int off = 16; off > 0; off >>= 1) s += __shfl_xor(s, off, 32);
        if (lane == 0) red[0] = s;
    }
    __syncthreads();
    const float inv = 1.0f / red[0];
#pragma unroll
    for (int i = 0; i < 8; i++) {
        const int j = i * 1024 + t;
        att[c * NGENES + j] = v[i] * inv;
    }
}

// ---------------------------------------------------------------- h_prime ---
// Grid: NCELLS*64 blocks of 128 threads; each block reduces 128 N-rows and
// atomically accumulates a 128-wide partial into hp[c,:] (pre-zeroed).
__global__ __launch_bounds__(128)
void gat_hprime(const float* __restrict__ hW, const float* __restrict__ att,
                float* __restrict__ hp) {
    const int c     = blockIdx.x >> 6;
    const int chunk = blockIdx.x & 63;
    const int f     = threadIdx.x;
    const float* hbase = hW + ((long)c * NGENES + chunk * 128) * OUT_F;
    const float* abase = att + c * NGENES + chunk * 128;
    float acc = 0.0f;
#pragma unroll 4
    for (int n = 0; n < 128; n++)
        acc = fmaf(abase[n], hbase[(long)n * OUT_F + f], acc);
    atomicAdd(&hp[c * OUT_F + f], acc);
}

// ----------------------------------------------------------------- launch ---
extern "C" void kernel_launch(void* const* d_in, const int* in_sizes, int n_in,
                              void* d_out, int out_size, void* d_ws, size_t ws_size,
                              hipStream_t stream) {
    const float* x   = (const float*)d_in[0];   // (32, 8192, 256) f32
    const int*   adj = (const int*)d_in[1];     // (32, 1, 8192) i32
    const float* W   = (const float*)d_in[2];   // (256, 128) f32
    const float* a   = (const float*)d_in[3];   // (32, 256) f32

    float* out = (float*)d_out;
    float* hp  = out;                           // (32,1,128)  = 4096 f32
    float* att = out + NCELLS * OUT_F;          // (32,1,8192) = 262144 f32

    char* ws = (char*)d_ws;
    _Float16* Wt = (_Float16*)ws;                                   // 64 KB
    float* hW = (float*)(ws + 65536);                               // 128 MB
    float* s1 = (float*)(ws + 65536 + (size_t)NCELLS * NGENES * OUT_F * 4);
    float* s2 = s1 + NCELLS * NGENES;

    (void)in_sizes; (void)n_in; (void)out_size; (void)ws_size;

    hipMemsetAsync(hp, 0, NCELLS * OUT_F * sizeof(float), stream);
    gat_prep_wt<<<128, 256, 0, stream>>>(W, Wt);
    gat_gemm<<<NCELLS * 32, 256, 65536, stream>>>(x, Wt, hW);
    gat_scores<<<NCELLS * NGENES / 8, 256, 0, stream>>>(hW, a, s1, s2);
    gat_softmax<<<NCELLS, 1024, 0, stream>>>(s1, s2, adj, att);
    gat_hprime<<<NCELLS * 64, 128, 0, stream>>>(hW, att, hp);
}
